// MuonLoRALinear_44014824849819
// MI455X (gfx1250) — matrix operand order repaired and checked by reading, not compiled, not run on hardware
//
#include <hip/hip_runtime.h>
#include <hip/hip_bf16.h>
#include <math.h>

// ---------------------------------------------------------------------------
// MuonLoRALinear for MI455X (gfx1250, wave32, WMMA + async global->LDS).
// out = x@Wt + bias + alpha * x@Mt,  M = NewtonSchulz5(B@A / ||B@A||)
// Heavy GEMMs: v_wmma_f32_16x16x32_bf16, LDS double-buffered via
// global_load_async_to_lds_b128 (ASYNCcnt); branch-free steady-state
// pipeline; epilogues specialized at compile time.
// Requires K % 64 == 0 and K >= 128 (true for all launches here: K=4096).
// ---------------------------------------------------------------------------

typedef __attribute__((ext_vector_type(16))) __bf16 bf16x16;
typedef __attribute__((ext_vector_type(8)))  float  f32x8;

#define TOKENS 8192
#define DIM    4096
#define RANK   16
#define NS_A   3.4445f
#define NS_B   (-4.775f)
#define NS_C   2.0315f
#define NS_EPS 1e-7f

// ---------------------------------------------------------------------------
// NT GEMM: C[m,n] = f( alphaAcc * sum_k P[m,k]*Q[n,k] )
//   MODE 0: C(bf16) = alphaAcc*acc
//   MODE 1: C(bf16) = alphaAcc*acc + alphaE*E[m,n]
//   MODE 2: C(f32)  = alphaAcc*acc + bias[n]
// P: [M,K] bf16 row-major, Q: [N,K] bf16 row-major. M%128==0, N%128==0.
// Block = 256 threads = 8 waves (4 m-waves x 2 n-waves); wave tile 32x64.
// ---------------------------------------------------------------------------
template <int MODE>
__global__ __launch_bounds__(256)
void gemm_nt_bf16(const __bf16* __restrict__ P, const __bf16* __restrict__ Q,
                  void* __restrict__ Cout, const __bf16* __restrict__ E,
                  const float* __restrict__ bias,
                  int M, int N, int K, float alphaAcc, float alphaE)
{
    constexpr int BM = 128, BN = 128, BK = 32;
    constexpr int LDT = 40; // bf16 per LDS row: 32 data + 8 pad (80B, 16B aligned)
    constexpr unsigned BUFB = (unsigned)(BM * LDT * sizeof(__bf16)); // 10240 B
    __shared__ __align__(16) __bf16 ldsA[2][BM * LDT];
    __shared__ __align__(16) __bf16 ldsB[2][BN * LDT];

    const int t    = threadIdx.x;
    const int lane = t & 31;
    const int wid  = t >> 5;
    const int wm   = (wid & 3) * 32;   // wave m-offset inside block tile
    const int wn   = (wid >> 2) * 64;  // wave n-offset inside block tile
    const int lrow = lane & 15;
    const int g    = lane >> 4;        // lane half: selects K sub-blocks
    const int m0   = blockIdx.y * BM;
    const int n0   = blockIdx.x * BN;

    // --- staging geometry: thread t owns rows (t>>2) and (t>>2)+64, 16B k-chunk
    const int srow = t >> 2;          // 0..63
    const int skc  = (t & 3) * 8;     // bf16 offset (8 bf16 = 16B)

    const unsigned la0 = (unsigned)(uintptr_t)(&ldsA[0][srow * LDT + skc]);
    const unsigned la1 = (unsigned)(uintptr_t)(&ldsA[0][(srow + 64) * LDT + skc]);
    const unsigned lb0 = (unsigned)(uintptr_t)(&ldsB[0][srow * LDT + skc]);
    const unsigned lb1 = (unsigned)(uintptr_t)(&ldsB[0][(srow + 64) * LDT + skc]);

    const __bf16* gA0 = P + (size_t)(m0 + srow) * K + skc;
    const __bf16* gA1 = P + (size_t)(m0 + srow + 64) * K + skc;
    const __bf16* gB0 = Q + (size_t)(n0 + srow) * K + skc;
    const __bf16* gB1 = Q + (size_t)(n0 + srow + 64) * K + skc;

    // Async stage of one 128x32 A tile + 128x32 B tile into buffer `buf`.
    auto stage = [&](int buf, int k0) {
        const unsigned o = buf ? BUFB : 0u;
        asm volatile(
            "global_load_async_to_lds_b128 %0, %4, off\n\t"
            "global_load_async_to_lds_b128 %1, %5, off\n\t"
            "global_load_async_to_lds_b128 %2, %6, off\n\t"
            "global_load_async_to_lds_b128 %3, %7, off"
            :: "v"(la0 + o), "v"(la1 + o), "v"(lb0 + o), "v"(lb1 + o),
               "v"((unsigned long long)(uintptr_t)(gA0 + k0)),
               "v"((unsigned long long)(uintptr_t)(gA1 + k0)),
               "v"((unsigned long long)(uintptr_t)(gB0 + k0)),
               "v"((unsigned long long)(uintptr_t)(gB1 + k0))
            : "memory");
    };

    f32x8 acc[2][4];
#pragma unroll
    for (int mi = 0; mi < 2; ++mi)
#pragma unroll
        for (int nj = 0; nj < 4; ++nj) acc[mi][nj] = (f32x8){};

    // 8 WMMAs against LDS buffer `buf` (compile-time constant).
    auto compute = [&](int buf) {
        union { bf16x16 v; uint4 u[2]; } af[2], bfr[4];
#pragma unroll
        for (int mi = 0; mi < 2; ++mi) {
            const __bf16* ap = &ldsA[buf][(wm + 16 * mi + lrow) * LDT] + g * 8;
            af[mi].u[0] = *(const uint4*)(ap);
            af[mi].u[1] = *(const uint4*)(ap + 16);
        }
#pragma unroll
        for (int nj = 0; nj < 4; ++nj) {
            const __bf16* bp = &ldsB[buf][(wn + 16 * nj + lrow) * LDT] + g * 8;
            bfr[nj].u[0] = *(const uint4*)(bp);
            bfr[nj].u[1] = *(const uint4*)(bp + 16);
        }
#pragma unroll
        for (int mi = 0; mi < 2; ++mi)
#pragma unroll
            for (int nj = 0; nj < 4; ++nj)
                acc[mi][nj] = __builtin_amdgcn_wmma_f32_16x16x32_bf16(
                    false, af[mi].v, false, bfr[nj].v, (short)0, acc[mi][nj],
                    false, false);
    };

    // ---- software pipeline: both buffers in flight, straight-line body ----
    stage(0, 0);
    stage(1, BK);
    // Invariant at loop top: buf0 <- tile k0, buf1 <- tile k0+BK (in flight),
    // ASYNCcnt <= 8. Inside the loop k0+3*BK <= K-BK, so restages are valid.
    for (int k0 = 0; k0 < K - 2 * BK; k0 += 2 * BK) {
        asm volatile("s_wait_asynccnt 0x4" ::: "memory"); // buf0 landed
        __syncthreads();
        compute(0);
        __syncthreads();                                  // buf0 free to restage
        stage(0, k0 + 2 * BK);
        asm volatile("s_wait_asynccnt 0x4" ::: "memory"); // buf1 landed
        __syncthreads();
        compute(1);
        __syncthreads();                                  // buf1 free to restage
        stage(1, k0 + 3 * BK);
    }
    // ---- last pair (tiles K-2*BK, K-BK), no restaging ----
    asm volatile("s_wait_asynccnt 0x4" ::: "memory");
    __syncthreads();
    compute(0);
    asm volatile("s_wait_asynccnt 0x0" ::: "memory");
    __syncthreads();
    compute(1);

    // Epilogue. C/D layout: VGPR r -> M = r + 8*g, N = lane%16.
    const int colBase = n0 + wn + lrow;
    float bv[4];
    if constexpr (MODE == 2) {
#pragma unroll
        for (int nj = 0; nj < 4; ++nj) bv[nj] = bias[colBase + 16 * nj];
    }
#pragma unroll
    for (int mi = 0; mi < 2; ++mi) {
        const int rowBase = m0 + wm + 16 * mi + g * 8;
#pragma unroll
        for (int r = 0; r < 8; ++r) {
            const size_t rowOff = (size_t)(rowBase + r) * N + colBase;
#pragma unroll
            for (int nj = 0; nj < 4; ++nj) {
                float o = alphaAcc * acc[mi][nj][r];
                if constexpr (MODE == 1)
                    o = fmaf(alphaE, (float)E[rowOff + 16 * nj], o);
                if constexpr (MODE == 2)
                    ((float*)Cout)[rowOff + 16 * nj] = o + bv[nj];
                else
                    ((__bf16*)Cout)[rowOff + 16 * nj] = (__bf16)o;
            }
        }
    }
}

// ---------------------------------------------------------------------------
// G = lora_B @ lora_A (f32), plus per-block partial sums of G^2 (deterministic).
// ---------------------------------------------------------------------------
__global__ __launch_bounds__(256)
void lora_g_norm(const float* __restrict__ lB, const float* __restrict__ lA,
                 float* __restrict__ G, float* __restrict__ part)
{
    const int idx = blockIdx.x * 256 + threadIdx.x;
    const int o = idx >> 12;      // / DIM
    const int i = idx & (DIM - 1);
    float acc = 0.f;
#pragma unroll
    for (int r = 0; r < RANK; ++r)
        acc = fmaf(lB[o * RANK + r], lA[r * DIM + i], acc);
    G[idx] = acc;

    __shared__ float red[256];
    red[threadIdx.x] = acc * acc;
    __syncthreads();
    for (int s = 128; s > 0; s >>= 1) {
        if (threadIdx.x < s) red[threadIdx.x] += red[threadIdx.x + s];
        __syncthreads();
    }
    if (threadIdx.x == 0) part[blockIdx.x] = red[0];
}

// Single-block deterministic reduce: s[0] = alpha = ||G||, s[1] = 1/(alpha+eps)
__global__ __launch_bounds__(256)
void finish_alpha(const float* __restrict__ part, float* __restrict__ s, int n)
{
    __shared__ float red[256];
    float a = 0.f;
    for (int i = threadIdx.x; i < n; i += 256) a += part[i];
    red[threadIdx.x] = a;
    __syncthreads();
    for (int st = 128; st > 0; st >>= 1) {
        if (threadIdx.x < st) red[threadIdx.x] += red[threadIdx.x + st];
        __syncthreads();
    }
    if (threadIdx.x == 0) {
        float al = sqrtf(red[0]);
        s[0] = al;
        s[1] = 1.f / (al + NS_EPS);
    }
}

// X0 = bf16(G * (*scale))
__global__ __launch_bounds__(256)
void scale_to_bf16(const float* __restrict__ in, const float* __restrict__ scale,
                   __bf16* __restrict__ out)
{
    const int i = blockIdx.x * 256 + threadIdx.x;
    out[i] = (__bf16)(in[i] * scale[0]);
}

// Wc = bf16(W + alpha * Xf)   (alpha = s[0])
__global__ __launch_bounds__(256)
void combine_w(const float* __restrict__ W, const __bf16* __restrict__ Xf,
               const float* __restrict__ s, __bf16* __restrict__ out)
{
    const int i = blockIdx.x * 256 + threadIdx.x;
    out[i] = (__bf16)(fmaf(s[0], (float)Xf[i], W[i]));
}

__global__ __launch_bounds__(256)
void cvt_f32_bf16(const float* __restrict__ in, __bf16* __restrict__ out)
{
    const int i = blockIdx.x * 256 + threadIdx.x;
    out[i] = (__bf16)in[i];
}

// 32x32 tiled bf16 transpose (square n x n)
__global__ __launch_bounds__(256)
void transpose_bf16(const __bf16* __restrict__ in, __bf16* __restrict__ out, int n)
{
    __shared__ __bf16 tile[32][33];
    const int bx = blockIdx.x * 32, by = blockIdx.y * 32;
    const int tx = threadIdx.x & 31, ty = threadIdx.x >> 5;
    for (int r = ty; r < 32; r += 8)
        tile[r][tx] = in[(size_t)(by + r) * n + bx + tx];
    __syncthreads();
    for (int r = ty; r < 32; r += 8)
        out[(size_t)(bx + r) * n + by + tx] = tile[tx][r];
}

// ---------------------------------------------------------------------------
extern "C" void kernel_launch(void* const* d_in, const int* in_sizes, int n_in,
                              void* d_out, int out_size, void* d_ws, size_t ws_size,
                              hipStream_t stream)
{
    (void)in_sizes; (void)n_in; (void)out_size; (void)ws_size;
    const float* x  = (const float*)d_in[0];   // [8192, 4096]
    const float* W  = (const float*)d_in[1];   // [4096, 4096]
    const float* bi = (const float*)d_in[2];   // [4096]
    const float* lA = (const float*)d_in[3];   // [16, 4096]
    const float* lB = (const float*)d_in[4];   // [4096, 16]

    constexpr size_t NN   = (size_t)DIM * DIM;           // 16,777,216
    constexpr size_t XN   = (size_t)TOKENS * DIM;        // 33,554,432
    constexpr int    GB_N = (int)(NN / 256);             // G blocks = 65536

    // Workspace carve-up (G f32 region later reused as xb bf16: both 64 MiB)
    char*   w    = (char*)d_ws;
    float*  s    = (float*)w;                       size_t off = 256;
    float*  part = (float*)(w + off);               off += (size_t)GB_N * 4;
    float*  G    = (float*)(w + off);
    __bf16* xb   = (__bf16*)(w + off);              off += NN * 4;   // aliased, disjoint lifetimes
    __bf16* X0   = (__bf16*)(w + off);              off += NN * 2;
    __bf16* X1   = (__bf16*)(w + off);              off += NN * 2;
    __bf16* Xt   = (__bf16*)(w + off);              off += NN * 2;
    __bf16* Am   = (__bf16*)(w + off);              off += NN * 2;
    __bf16* Bm   = (__bf16*)(w + off);              off += NN * 2;
    __bf16* Wc   = (__bf16*)(w + off);              off += NN * 2;

    const dim3 blk(256);
    const dim3 gSq(DIM / 128, DIM / 128);     // (32, 32)  4096x4096 GEMM
    const dim3 gBig(DIM / 128, TOKENS / 128); // (32, 64)  8192x4096 GEMM
    const dim3 gTr(DIM / 32, DIM / 32);       // (128,128)

    // 1) G = lora_B @ lora_A ; partial ||G||^2
    lora_g_norm<<<GB_N, blk, 0, stream>>>(lB, lA, G, part);
    // 2) alpha, 1/(alpha+eps)
    finish_alpha<<<1, blk, 0, stream>>>(part, s, GB_N);
    // 3) X0 = bf16(G / (alpha+eps)); Xt = X0^T
    scale_to_bf16<<<(int)(NN / 256), blk, 0, stream>>>(G, s + 1, X0);
    transpose_bf16<<<gTr, blk, 0, stream>>>(X0, Xt, DIM);

    // 4) Newton-Schulz (bf16 data, f32 WMMA accumulate)
    for (int i = 0; i < 5; ++i) {
        const __bf16* Xs = (i & 1) ? X1 : X0;
        __bf16*       Xd = (i & 1) ? X0 : X1;
        // A = X X^T                         (NT)
        gemm_nt_bf16<0><<<gSq, blk, 0, stream>>>(Xs, Xs, Am, nullptr, nullptr,
                                                 DIM, DIM, DIM, 1.f, 0.f);
        // Bm = c*(A@A) + b*A  (A symmetric -> A@A == NT(A,A))
        gemm_nt_bf16<1><<<gSq, blk, 0, stream>>>(Am, Am, Bm, Am, nullptr,
                                                 DIM, DIM, DIM, NS_C, NS_B);
        // Xd = Bm@X + a*X  ==  NT(Bm, X^T) + a*X
        gemm_nt_bf16<1><<<gSq, blk, 0, stream>>>(Bm, Xt, Xd, Xs, nullptr,
                                                 DIM, DIM, DIM, 1.f, NS_A);
        if (i < 4) transpose_bf16<<<gTr, blk, 0, stream>>>(Xd, Xt, DIM);
    }
    // After 5 iters the result lives in X1.

    // 5) Fused weight: Wc = bf16(W + alpha * M)
    combine_w<<<(int)(NN / 256), blk, 0, stream>>>(W, X1, s, Wc);
    // 6) xb = bf16(x)   (reuses G region; G is dead)
    cvt_f32_bf16<<<(int)(XN / 256), blk, 0, stream>>>(x, xb);
    // 7) out = x @ Wc^T + bias   (single big GEMM, f32 out)
    gemm_nt_bf16<2><<<gBig, blk, 0, stream>>>(xb, Wc, d_out, nullptr, bi,
                                              TOKENS, DIM, DIM, 1.f, 0.f);
}
